// LSTM_11802570129824
// MI455X (gfx1250) — matrix operand order repaired and checked by reading, not compiled, not run on hardware
//
#include <hip/hip_runtime.h>
#include <hip/hip_bf16.h>
#include <math.h>

// Shapes from the reference
#define T_STEPS 512
#define D_IN    128
#define H_DIM   128
#define G_DIM   512   // 4*H
#define O_DIM   64

typedef __attribute__((ext_vector_type(16))) _Float16 v16h;
typedef __attribute__((ext_vector_type(8)))  _Float16 v8h;
typedef __attribute__((ext_vector_type(8)))  float    v8f;

union V16H { v16h v; _Float16 e[16]; };
union V8H  { v8h  v; _Float16 e[8];  };
union V8F  { v8f  v; float    e[8];  };

__device__ __forceinline__ float sigmoid_f(float x) {
  return 1.0f / (1.0f + __expf(-x));
}

// ---------------------------------------------------------------------------
// Phase 1: xg[t,g] = sum_d x[255,t,d] * W_ih[g,d] + b_ih[g] + b_hh[g]
// 512x512 output; one 16x16 tile per wave; K=128 -> 4 wmma f16 steps.
// ---------------------------------------------------------------------------
__global__ __launch_bounds__(256) void xg_gemm_kernel(
    const float* __restrict__ x,
    const float* __restrict__ W_ih,
    const float* __restrict__ b_ih,
    const float* __restrict__ b_hh,
    float* __restrict__ xg)
{
  const int lane = threadIdx.x & 31;
  const int wave = threadIdx.x >> 5;
  const int l16  = lane & 15;
  const int half = lane >> 4;
  const int tile = blockIdx.x * 8 + wave;   // 0..1023
  const int tM   = tile >> 5;               // t-tile 0..31
  const int tN   = tile & 31;               // g-tile 0..31

  const float* x255 = x + (size_t)255 * T_STEPS * D_IN;
  const int trow = tM * 16 + l16;           // A row (timestep)
  const int g    = tN * 16 + l16;           // B col (gate)

  v8f acc = {};
#pragma unroll
  for (int kk = 0; kk < 4; ++kk) {
    // A operand: 16x32 f16.  K = (e/8)*16 + half*8 + e%8  (+ kk*32)
    V16H a;
    const float* ap = x255 + (size_t)trow * D_IN + kk * 32 + half * 8;
#pragma unroll
    for (int i = 0; i < 8; ++i) {
      a.e[i]     = (_Float16)ap[i];
      a.e[i + 8] = (_Float16)ap[i + 16];
    }
    // B operand: 32x16 f16.  K = (e/8)*16 + half*8 + e%8  (+ kk*32), N = l16.
    // B[k][n] = W_ih[n][k]; two runs of 8 floats along d, 16 apart.
    V16H b;
    const float* bp = W_ih + (size_t)g * D_IN + kk * 32 + half * 8;
    __builtin_prefetch((const void*)(bp + 32), 0, 0);
#pragma unroll
    for (int i = 0; i < 16; ++i) b.e[i] = (_Float16)bp[i + (i & 8)];

    acc = __builtin_amdgcn_wmma_f32_16x16x32_f16(
        false, a.v, false, b.v, (short)0, acc, false, false);
  }

  V8F av; av.v = acc;
  const float bias = b_ih[g] + b_hh[g];   // bias depends only on N = l16 col
#pragma unroll
  for (int r = 0; r < 8; ++r) {
    const int t = tM * 16 + r + 8 * half;   // C/D layout: M = r + 8*half
    xg[(size_t)t * G_DIM + g] = av.e[r] + bias;
  }
}

// ---------------------------------------------------------------------------
// Phase 2: serial LSTM scan for the single live sequence (batch 255).
// One workgroup, 8 waves.  W_hh^T lives entirely in VGPRs (B-layout tiles),
// preloaded once; the 512-step loop touches only LDS.
// The xg row for step t+1 is double-buffered into LDS with the CDNA5 async
// global->LDS DMA path (ASYNCcnt), issued at the top of step t so the L2
// round-trip overlaps the entire step's compute and barriers.
// ---------------------------------------------------------------------------
__global__ __launch_bounds__(256) void lstm_scan_kernel(
    const float* __restrict__ xg,     // (T, 4H)
    const float* __restrict__ W_hh,   // (4H, H)
    _Float16* __restrict__ h_hist)    // (256, 128): h[255, 256+j]
{
  __shared__ __align__(16) _Float16 h_lds[H_DIM];
  __shared__ float gates_lds[G_DIM];
  __shared__ __align__(16) float xg_buf[2][G_DIM];

  const int tid  = threadIdx.x;
  const int lane = tid & 31;
  const int wave = tid >> 5;
  const int l16  = lane & 15;
  const int half = lane >> 4;

  // Preload B = W_hh^T: wave handles n-tiles (wave*4 .. wave*4+3), 4 k-tiles.
  V16H breg[4][4];
#pragma unroll
  for (int nt = 0; nt < 4; ++nt) {
    const int g = (wave * 4 + nt) * 16 + l16;
#pragma unroll
    for (int kt = 0; kt < 4; ++kt) {
      const float* bp = W_hh + (size_t)g * H_DIM + kt * 32 + half * 8;
#pragma unroll
      for (int i = 0; i < 16; ++i) breg[nt][kt].e[i] = (_Float16)bp[i + (i & 8)];
    }
  }

  if (tid < H_DIM) h_lds[tid] = (_Float16)0.0f;
  float c_state = 0.0f;

  // Prologue: async-copy xg row 0 into xg_buf[0] (256 lanes x 8B = 2KB row).
  {
    const unsigned dst =
        (unsigned)(uintptr_t)(&xg_buf[0][tid * 2]);            // LDS byte addr
    const unsigned long long src =
        (unsigned long long)(uintptr_t)(xg + (size_t)tid * 2); // global addr
    asm volatile("global_load_async_to_lds_b64 %0, %1, off"
                 :: "v"(dst), "v"(src) : "memory");
    asm volatile("s_wait_asynccnt 0x0" ::: "memory");
  }
  __syncthreads();   // row 0 + h_lds init visible to all waves

  for (int t = 0; t < T_STEPS; ++t) {
    // Issue async DMA of xg row t+1 now; it drains at this step's trailing
    // barrier, hiding the L2 latency behind the whole step's compute.
    if (t + 1 < T_STEPS) {
      const unsigned dst =
          (unsigned)(uintptr_t)(&xg_buf[(t + 1) & 1][tid * 2]);
      const unsigned long long src = (unsigned long long)(uintptr_t)(
          xg + (size_t)(t + 1) * G_DIM + tid * 2);
      asm volatile("global_load_async_to_lds_b64 %0, %1, off"
                   :: "v"(dst), "v"(src) : "memory");
    }

    // A tiles: only M==0 (lanes with l16==0) carry h; others are zero.
    V16H a[4];
#pragma unroll
    for (int kt = 0; kt < 4; ++kt) {
      if (l16 == 0) {
        V8H lo, hi;
        lo.v = *(const v8h*)(h_lds + kt * 32 + half * 8);
        hi.v = *(const v8h*)(h_lds + kt * 32 + 16 + half * 8);
#pragma unroll
        for (int i = 0; i < 8; ++i) { a[kt].e[i] = lo.e[i]; a[kt].e[i + 8] = hi.e[i]; }
      } else {
#pragma unroll
        for (int i = 0; i < 16; ++i) a[kt].e[i] = (_Float16)0.0f;
      }
    }

    const float* xg_row = xg_buf[t & 1];
#pragma unroll
    for (int nt = 0; nt < 4; ++nt) {
      v8f acc = {};
#pragma unroll
      for (int kt = 0; kt < 4; ++kt)
        acc = __builtin_amdgcn_wmma_f32_16x16x32_f16(
            false, a[kt].v, false, breg[nt][kt].v, (short)0, acc, false, false);
      V8F av; av.v = acc;
      const int g = (wave * 4 + nt) * 16 + l16;         // gate index
      const float pre = av.e[0] + xg_row[g];
      const int gate_type = g >> 7;                      // i,f,g,o (uniform/tile)
      const float act = (gate_type == 2) ? tanhf(pre) : sigmoid_f(pre);
      if (lane < 16) gates_lds[g] = act;                 // row M=0 lives in lanes 0..15
    }
    __syncthreads();

    if (tid < H_DIM) {
      const float i_g = gates_lds[tid];
      const float f_g = gates_lds[tid + 128];
      const float g_g = gates_lds[tid + 256];
      const float o_g = gates_lds[tid + 384];
      c_state = f_g * c_state + i_g * g_g;
      const float h = o_g * tanhf(c_state);
      const _Float16 hh = (_Float16)h;
      h_lds[tid] = hh;
      if (t >= 256) h_hist[(size_t)(t - 256) * H_DIM + tid] = hh;
    }
    // Drain this wave's async DMA (row t+1), then the trailing barrier makes
    // both h_lds and xg_buf[(t+1)&1] visible to every wave.
    asm volatile("s_wait_asynccnt 0x0" ::: "memory");
    __syncthreads();
  }
}

// ---------------------------------------------------------------------------
// Phase 3: out = (h_hist @ W1^T + b1) @ W2^T + b2, fused per 16-row tile.
// 16 blocks x 8 waves.  Layer-1 result staged through LDS (f16, A-layout).
// ---------------------------------------------------------------------------
__global__ __launch_bounds__(256) void mlp_kernel(
    const _Float16* __restrict__ h_hist,  // (256,128) f16
    const float* __restrict__ W1, const float* __restrict__ b1,
    const float* __restrict__ W2, const float* __restrict__ b2,
    float* __restrict__ out)              // (256,64) f32
{
  __shared__ __align__(16) _Float16 h1_lds[16 * H_DIM];

  const int lane = threadIdx.x & 31;
  const int wave = threadIdx.x >> 5;
  const int l16  = lane & 15;
  const int half = lane >> 4;
  const int jt   = blockIdx.x;  // 0..15 row tile

  // ---- Layer 1: each wave computes one 16x16 N-tile of h1 (N = wave*16..)
  {
    const int n1 = wave * 16 + l16;
    v8f acc = {};
#pragma unroll
    for (int kk = 0; kk < 4; ++kk) {
      V16H a;
      const _Float16* ap = h_hist + (size_t)(jt * 16 + l16) * H_DIM + kk * 32 + half * 8;
      V8H lo, hi;
      lo.v = *(const v8h*)(ap);
      hi.v = *(const v8h*)(ap + 16);
#pragma unroll
      for (int i = 0; i < 8; ++i) { a.e[i] = lo.e[i]; a.e[i + 8] = hi.e[i]; }

      V16H b;
      const float* bp = W1 + (size_t)n1 * H_DIM + kk * 32 + half * 8;
      __builtin_prefetch((const void*)(bp + 32), 0, 0);
#pragma unroll
      for (int i = 0; i < 16; ++i) b.e[i] = (_Float16)bp[i + (i & 8)];

      acc = __builtin_amdgcn_wmma_f32_16x16x32_f16(
          false, a.v, false, b.v, (short)0, acc, false, false);
    }
    V8F av; av.v = acc;
    const float bias = b1[n1];
#pragma unroll
    for (int r = 0; r < 8; ++r) {
      const int row = r + 8 * half;                 // local row 0..15
      h1_lds[row * H_DIM + n1] = (_Float16)(av.e[r] + bias);
    }
  }
  __syncthreads();

  // ---- Layer 2: waves 0..3 compute the 4 N-tiles of O=64 (uniform branch).
  if (wave < 4) {
    const int n2 = wave * 16 + l16;
    v8f acc = {};
#pragma unroll
    for (int kk = 0; kk < 4; ++kk) {
      V16H a;
      const _Float16* ap = h1_lds + (size_t)l16 * H_DIM + kk * 32 + half * 8;
      V8H lo, hi;
      lo.v = *(const v8h*)(ap);
      hi.v = *(const v8h*)(ap + 16);
#pragma unroll
      for (int i = 0; i < 8; ++i) { a.e[i] = lo.e[i]; a.e[i + 8] = hi.e[i]; }

      V16H b;
      const float* bp = W2 + (size_t)n2 * H_DIM + kk * 32 + half * 8;
#pragma unroll
      for (int i = 0; i < 16; ++i) b.e[i] = (_Float16)bp[i + (i & 8)];

      acc = __builtin_amdgcn_wmma_f32_16x16x32_f16(
          false, a.v, false, b.v, (short)0, acc, false, false);
    }
    V8F av; av.v = acc;
    const float bias = b2[n2];
#pragma unroll
    for (int r = 0; r < 8; ++r) {
      const int j = jt * 16 + r + 8 * half;
      out[(size_t)j * O_DIM + n2] = av.e[r] + bias;
    }
  }
}

// ---------------------------------------------------------------------------
extern "C" void kernel_launch(void* const* d_in, const int* in_sizes, int n_in,
                              void* d_out, int out_size, void* d_ws, size_t ws_size,
                              hipStream_t stream) {
  const float* x    = (const float*)d_in[0];
  const float* W_ih = (const float*)d_in[1];
  const float* W_hh = (const float*)d_in[2];
  const float* b_ih = (const float*)d_in[3];
  const float* b_hh = (const float*)d_in[4];
  const float* W1   = (const float*)d_in[5];
  const float* b1   = (const float*)d_in[6];
  const float* W2   = (const float*)d_in[7];
  const float* b2   = (const float*)d_in[8];
  float* out = (float*)d_out;

  // Workspace layout: xg (512x512 f32 = 1 MiB), then h_hist (256x128 f16).
  float*    xg     = (float*)d_ws;
  _Float16* h_hist = (_Float16*)((char*)d_ws + (size_t)T_STEPS * G_DIM * sizeof(float));

  xg_gemm_kernel<<<128, 256, 0, stream>>>(x, W_ih, b_ih, b_hh, xg);
  lstm_scan_kernel<<<1, 256, 0, stream>>>(xg, W_hh, h_hist);
  mlp_kernel<<<16, 256, 0, stream>>>(h_hist, W1, b1, W2, b2, out);
}
